// BitMambaBlock_88639535055169
// MI455X (gfx1250) — 8-bit matrix operand order repaired and checked by reading, compiled, not run on hardware
//
#include <hip/hip_runtime.h>
#include <stdint.h>

#define DIM        2048
#define D_STATE    128
#define D_CONV     4
#define HEADDIM    64
#define D_INNER    4096
#define NHEADS     64
#define D_IN_PROJ  8512
#define CONV_DIM   4352
#define BATCH      2
#define SEQLEN     2048
#define ROWS       (BATCH*SEQLEN)

typedef __attribute__((ext_vector_type(8))) int v8i;

__device__ __forceinline__ float block_reduce_sum(float v, float* sdata) {
    int t = threadIdx.x;
    sdata[t] = v; __syncthreads();
    for (int s = 128; s > 0; s >>= 1) { if (t < s) sdata[t] += sdata[t + s]; __syncthreads(); }
    float r = sdata[0]; __syncthreads();
    return r;
}
__device__ __forceinline__ float block_reduce_max(float v, float* sdata) {
    int t = threadIdx.x;
    sdata[t] = v; __syncthreads();
    for (int s = 128; s > 0; s >>= 1) { if (t < s) sdata[t] = fmaxf(sdata[t], sdata[t + s]); __syncthreads(); }
    float r = sdata[0]; __syncthreads();
    return r;
}

__global__ void zero_red_kernel(float* red) {
    if (threadIdx.x < 2) red[threadIdx.x] = 0.0f;
}

__global__ void absmean_kernel(const float* __restrict__ w, size_t n, float* accum) {
    __shared__ float sdata[256];
    float s = 0.0f;
    for (size_t i = (size_t)blockIdx.x * blockDim.x + threadIdx.x; i < n;
         i += (size_t)gridDim.x * blockDim.x)
        s += fabsf(w[i]);
    s = block_reduce_sum(s, sdata);
    if (threadIdx.x == 0) atomicAdd(accum, s);
}

__global__ void wquant_kernel(const float* __restrict__ w, size_t n,
                              const float* __restrict__ accum, int8_t* __restrict__ q) {
    float scale = fmaxf(accum[0] / (float)n, 1e-5f);
    float inv = 1.0f / scale;
    for (size_t i = (size_t)blockIdx.x * blockDim.x + threadIdx.x; i < n;
         i += (size_t)gridDim.x * blockDim.x) {
        float ws = w[i] * inv;
        float r  = (ws >= 0.0f) ? floorf(ws + 0.5f) : ceilf(ws - 0.5f);
        r = fminf(fmaxf(r, -1.0f), 1.0f);
        q[i] = (int8_t)(int)r;
    }
}

#define MAXE 16
__global__ void rms_ln_quant_kernel(const float* __restrict__ x, const float* __restrict__ w,
                                    int8_t* __restrict__ q, float* __restrict__ ascale, int D) {
    __shared__ float sdata[256];
    int row = blockIdx.x, t = threadIdx.x;
    int per = D >> 8;
    const float* xr = x + (size_t)row * D;
    float v[MAXE];

    float ss = 0.0f;
    for (int i = 0; i < per; ++i) { v[i] = xr[i * 256 + t]; ss += v[i] * v[i]; }
    ss = block_reduce_sum(ss, sdata);
    float rinv = rsqrtf(ss / (float)D + 1e-6f);
    for (int i = 0; i < per; ++i) v[i] = w[i * 256 + t] * (v[i] * rinv);

    float m = 0.0f;
    for (int i = 0; i < per; ++i) m += v[i];
    m = block_reduce_sum(m, sdata) / (float)D;
    float var = 0.0f;
    for (int i = 0; i < per; ++i) { float d = v[i] - m; var += d * d; }
    var = block_reduce_sum(var, sdata) / (float)D;
    float lrinv = rsqrtf(var + 1e-5f);
    float amax = 0.0f;
    for (int i = 0; i < per; ++i) { v[i] = (v[i] - m) * lrinv; amax = fmaxf(amax, fabsf(v[i])); }
    amax = fmaxf(block_reduce_max(amax, sdata), 1e-5f);

    float scale = 127.0f / amax;
    int8_t* qr = q + (size_t)row * D;
    for (int i = 0; i < per; ++i) {
        float qv = rintf(v[i] * scale);
        qv = fminf(fmaxf(qv, -128.0f), 127.0f);
        qr[i * 256 + t] = (int8_t)(int)qv;
    }
    if (t == 0) ascale[row] = amax / 127.0f;
}

template<int K>
__global__ void
__launch_bounds__(256, 2)
gemm_iu8_wmma_kernel(const int8_t* __restrict__ A, const int8_t* __restrict__ Bw,
                     const float* __restrict__ ascale,
                     const float* __restrict__ residual,
                     float* __restrict__ out,
                     int M, int N, int ldOut) {
    int wave = blockIdx.x * (blockDim.x >> 5) + (threadIdx.x >> 5);
    int lane = threadIdx.x & 31;
    int n64  = N >> 6;
    int mt   = wave / n64;
    int nt   = wave - mt * n64;
    if (mt >= (M >> 4)) return;
    int m0 = mt << 4, n0 = nt << 6;
    int half = lane >> 4, r = lane & 15;

    const int8_t* arow = A + (size_t)(m0 + r) * K + half * 16;
    const int8_t* brow = Bw + (size_t)(n0 + r) * K + half * 16;

    v8i acc0 = {}, acc1 = {}, acc2 = {}, acc3 = {};

    #define LOAD_A(af, kb)                                                    \
        {   int2 t0 = *(const int2*)(arow + (kb) +  0);                       \
            int2 t1 = *(const int2*)(arow + (kb) +  8);                       \
            int2 t2 = *(const int2*)(arow + (kb) + 32);                       \
            int2 t3 = *(const int2*)(arow + (kb) + 40);                       \
            af[0]=t0.x; af[1]=t0.y; af[2]=t1.x; af[3]=t1.y;                   \
            af[4]=t2.x; af[5]=t2.y; af[6]=t3.x; af[7]=t3.y; }
    #define LOAD_B(bf, j, kb)                                                 \
        {   int4 q0 = *(const int4*)(brow + (size_t)(j)*16*K + (kb));         \
            int4 q1 = *(const int4*)(brow + (size_t)(j)*16*K + (kb) + 32);    \
            bf[0]=q0.x; bf[1]=q0.y; bf[2]=q0.z; bf[3]=q0.w;                   \
            bf[4]=q1.x; bf[5]=q1.y; bf[6]=q1.z; bf[7]=q1.w; }

    v8i af, bf0, bf1;
    LOAD_A(af, 0);
    LOAD_B(bf0, 0, 0); LOAD_B(bf1, 1, 0);

    for (int kb = 0; kb < K - 64; kb += 64) {
        v8i bf2, bf3;
        LOAD_B(bf2, 2, kb); LOAD_B(bf3, 3, kb);
        v8i an;
        LOAD_A(an, kb + 64);

        acc0 = __builtin_amdgcn_wmma_i32_16x16x64_iu8(true, af, true, bf0, acc0, false, false);
        acc1 = __builtin_amdgcn_wmma_i32_16x16x64_iu8(true, af, true, bf1, acc1, false, false);

        v8i b0n, b1n;
        LOAD_B(b0n, 0, kb + 64); LOAD_B(b1n, 1, kb + 64);

        acc2 = __builtin_amdgcn_wmma_i32_16x16x64_iu8(true, af, true, bf2, acc2, false, false);
        acc3 = __builtin_amdgcn_wmma_i32_16x16x64_iu8(true, af, true, bf3, acc3, false, false);

        af = an; bf0 = b0n; bf1 = b1n;
    }
    {
        v8i bf2, bf3;
        LOAD_B(bf2, 2, K - 64); LOAD_B(bf3, 3, K - 64);
        acc0 = __builtin_amdgcn_wmma_i32_16x16x64_iu8(true, af, true, bf0, acc0, false, false);
        acc1 = __builtin_amdgcn_wmma_i32_16x16x64_iu8(true, af, true, bf1, acc1, false, false);
        acc2 = __builtin_amdgcn_wmma_i32_16x16x64_iu8(true, af, true, bf2, acc2, false, false);
        acc3 = __builtin_amdgcn_wmma_i32_16x16x64_iu8(true, af, true, bf3, acc3, false, false);
    }
    #undef LOAD_A
    #undef LOAD_B

    int rowb = m0 + half * 8;
    float asc[8];
    for (int v = 0; v < 8; ++v) asc[v] = ascale[rowb + v];
    v8i accs[4] = { acc0, acc1, acc2, acc3 };
    for (int j = 0; j < 4; ++j) {
        int col = n0 + j * 16 + r;
        v8i a = accs[j];
        for (int v = 0; v < 8; ++v) {
            size_t idx = (size_t)(rowb + v) * ldOut + col;
            float val = (float)a[v] * asc[v];
            out[idx] = residual ? (residual[idx] + val) : val;
        }
    }
}

__global__ void conv_silu_kernel(const float* __restrict__ zxbcdt,
                                 const float* __restrict__ conv_w,
                                 const float* __restrict__ conv_b,
                                 float* __restrict__ xconv) {
    size_t total = (size_t)ROWS * CONV_DIM;
    for (size_t idx = (size_t)blockIdx.x * blockDim.x + threadIdx.x; idx < total;
         idx += (size_t)gridDim.x * blockDim.x) {
        int c  = (int)(idx % CONV_DIM);
        int bl = (int)(idx / CONV_DIM);
        int l  = bl & (SEQLEN - 1);
        int b  = bl >> 11;
        float acc = conv_b[c];
        #pragma unroll
        for (int j = 0; j < D_CONV; ++j) {
            int ll = l - (D_CONV - 1) + j;
            if (ll >= 0)
                acc += conv_w[c * D_CONV + j] *
                       zxbcdt[(size_t)(b * SEQLEN + ll) * D_IN_PROJ + D_INNER + c];
        }
        xconv[idx] = acc / (1.0f + expf(-acc));
    }
}

__global__ void ssd_scan_kernel(const float* __restrict__ zxbcdt,
                                const float* __restrict__ xconv,
                                const float* __restrict__ A_log,
                                const float* __restrict__ Dvec,
                                const float* __restrict__ dt_bias,
                                float* __restrict__ ybuf) {
    __shared__ float Bs[D_STATE];
    __shared__ float Cs[D_STATE];
    __shared__ float red[256];

    int blk = blockIdx.x;
    int b = blk >> 6, h = blk & 63;
    int t = threadIdx.x;
    int p = t & 63, ng = t >> 6;
    int nb = ng * 32;

    float Ah  = -expf(A_log[h]);
    float dtb = dt_bias[h];
    float Dh  = Dvec[h];

    float st[32];
    #pragma unroll
    for (int i = 0; i < 32; ++i) st[i] = 0.0f;

    for (int l = 0; l < SEQLEN; ++l) {
        size_t rb = (size_t)(b * SEQLEN + l);
        size_t xrow = rb * CONV_DIM;
        size_t zrow = rb * D_IN_PROJ;

        if (t < D_STATE)       Bs[t]           = xconv[xrow + D_INNER + t];
        else                   Cs[t - D_STATE] = xconv[xrow + D_INNER + D_STATE + (t - D_STATE)];
        float x_p = xconv[xrow + (size_t)h * HEADDIM + p];

        float dtr = zxbcdt[zrow + D_INNER + CONV_DIM + h] + dtb;
        float dt  = (dtr > 20.0f) ? dtr : log1pf(expf(dtr));
        float dA  = expf(dt * Ah);
        __syncthreads();

        float dtx = dt * x_p;
        float acc = 0.0f;
        #pragma unroll
        for (int i = 0; i < 32; ++i) {
            float s = st[i] * dA + dtx * Bs[nb + i];
            st[i] = s;
            acc += s * Cs[nb + i];
        }
        red[t] = acc;
        __syncthreads();

        if (t < HEADDIM) {
            float ys = red[t] + red[t + 64] + red[t + 128] + red[t + 192];
            float zv = zxbcdt[zrow + (size_t)h * HEADDIM + t];
            float sz = zv / (1.0f + expf(-zv));
            ybuf[rb * D_INNER + (size_t)h * HEADDIM + t] = (ys + Dh * x_p) * sz;
        }
    }
}

extern "C" void kernel_launch(void* const* d_in, const int* in_sizes, int n_in,
                              void* d_out, int out_size, void* d_ws, size_t ws_size,
                              hipStream_t stream) {
    const float* hidden     = (const float*)d_in[0];
    const float* in_proj_w  = (const float*)d_in[1];
    const float* out_proj_w = (const float*)d_in[2];
    const float* conv_w     = (const float*)d_in[3];
    const float* conv_b     = (const float*)d_in[4];
    const float* A_log      = (const float*)d_in[5];
    const float* Dvec       = (const float*)d_in[6];
    const float* dt_bias    = (const float*)d_in[7];
    const float* norm_w     = (const float*)d_in[8];
    const float* out_norm_w = (const float*)d_in[9];
    float* out = (float*)d_out;
    char* ws = (char*)d_ws;

    size_t o = 0;
    float*  red    = (float*)(ws + o);          o += 256;
    int8_t* wq1    = (int8_t*)(ws + o);         o += (size_t)D_IN_PROJ * DIM;
    int8_t* wq2    = (int8_t*)(ws + o);         o += (size_t)DIM * D_INNER;
    int8_t* aq     = (int8_t*)(ws + o);         o += (size_t)ROWS * D_INNER;
    float*  ascale = (float*)(ws + o);          o += (size_t)ROWS * sizeof(float);
    o = (o + 255) & ~(size_t)255;
    float*  zxbcdt = (float*)(ws + o);          o += (size_t)ROWS * D_IN_PROJ * 4;
    float*  xconv  = (float*)(ws + o);          o += (size_t)ROWS * CONV_DIM * 4;
    float*  ybuf   = (float*)(ws + o);          o += (size_t)ROWS * D_INNER * 4;

    const size_t nW1 = (size_t)D_IN_PROJ * DIM;
    const size_t nW2 = (size_t)DIM * D_INNER;

    zero_red_kernel<<<1, 32, 0, stream>>>(red);

    absmean_kernel<<<2048, 256, 0, stream>>>(in_proj_w,  nW1, red + 0);
    absmean_kernel<<<2048, 256, 0, stream>>>(out_proj_w, nW2, red + 1);
    wquant_kernel<<<4096, 256, 0, stream>>>(in_proj_w,  nW1, red + 0, wq1);
    wquant_kernel<<<4096, 256, 0, stream>>>(out_proj_w, nW2, red + 1, wq2);

    rms_ln_quant_kernel<<<ROWS, 256, 0, stream>>>(hidden, norm_w, aq, ascale, DIM);
    {
        int waves = (ROWS / 16) * (D_IN_PROJ / 64);
        gemm_iu8_wmma_kernel<DIM><<<(waves + 7) / 8, 256, 0, stream>>>(
            aq, wq1, ascale, nullptr, zxbcdt, ROWS, D_IN_PROJ, D_IN_PROJ);
    }

    {
        size_t total = (size_t)ROWS * CONV_DIM;
        conv_silu_kernel<<<(int)((total + 255) / 256), 256, 0, stream>>>(
            zxbcdt, conv_w, conv_b, xconv);
    }

    ssd_scan_kernel<<<BATCH * NHEADS, 256, 0, stream>>>(
        zxbcdt, xconv, A_log, Dvec, dt_bias, ybuf);

    rms_ln_quant_kernel<<<ROWS, 256, 0, stream>>>(ybuf, out_norm_w, aq, ascale, D_INNER);
    {
        int waves = (ROWS / 16) * (DIM / 64);
        gemm_iu8_wmma_kernel<D_INNER><<<(waves + 7) / 8, 256, 0, stream>>>(
            aq, wq2, ascale, hidden, out, ROWS, DIM, DIM);
    }
}
